// SlotMemoryCompressor_60352880443939
// MI455X (gfx1250) — matrix operand order repaired and checked by reading, not compiled, not run on hardware
//
#include <hip/hip_runtime.h>

typedef unsigned short ushort_t;
typedef __attribute__((ext_vector_type(16))) __bf16 v16bf;
typedef __attribute__((ext_vector_type(16))) unsigned short v16us;
typedef __attribute__((ext_vector_type(4))) unsigned int v4u;
typedef __attribute__((ext_vector_type(8))) float v8f;
typedef __attribute__((ext_vector_type(4))) float v4f;

union BFrag { v16us u; v16bf b; v4u q[2]; };

__device__ __forceinline__ ushort_t f2bf(float f) {
    unsigned int u = __float_as_uint(f);
    unsigned int r = (u + 0x7FFFu + ((u >> 16) & 1u)) >> 16;  // round-to-nearest-even
    return (ushort_t)r;
}
__device__ __forceinline__ float bf2f(ushort_t h) {
    return __uint_as_float(((unsigned int)h) << 16);
}
__device__ __forceinline__ v8f zero8() {
    v8f z;
#pragma unroll
    for (int i = 0; i < 8; ++i) z[i] = 0.f;
    return z;
}
__device__ __forceinline__ float gelu(float x) {
    return 0.5f * x * (1.f + erff(x * 0.70710678118654752f));
}

// ---- WMMA: D = A*B + C (bf16 in, f32 acc). hi/lo split-precision triple product.
__device__ __forceinline__ v8f wmma3(const BFrag& ah, const BFrag& al,
                                     const BFrag& bh, const BFrag& bl, v8f c) {
    c = __builtin_amdgcn_wmma_f32_16x16x32_bf16(false, ah.b, false, bh.b, (short)0, c, false, false);
    c = __builtin_amdgcn_wmma_f32_16x16x32_bf16(false, ah.b, false, bl.b, (short)0, c, false, false);
    c = __builtin_amdgcn_wmma_f32_16x16x32_bf16(false, al.b, false, bh.b, (short)0, c, false, false);
    return c;
}

// A fragment (16x32 bf16 tile, row-major source, stride in elems).
// Lane L: row = L&15, K runs [kb,kb+8) and [kb+16,kb+24), kb = (L<16)?0:8.
__device__ __forceinline__ void load_a_frag(const ushort_t* base, int stride, int lane, int k0, BFrag& f) {
    int r = lane & 15;
    int kb = (lane < 16) ? 0 : 8;
    const ushort_t* p = base + (size_t)r * stride + k0 + kb;
    f.q[0] = *(const v4u*)p;
    f.q[1] = *(const v4u*)(p + 16);
}

// B fragment from transposed weights [N][K]. Lane L: col = nBase + (L&15),
// K runs [kb,kb+8) and [kb+16,kb+24), kb = (L<16)?0:8.
__device__ __forceinline__ void load_b_frag(const ushort_t* baseT, int Kdim, int nBase, int lane, int k0, BFrag& f) {
    int col = nBase + (lane & 15);
    int kb = (lane < 16) ? 0 : 8;
    const ushort_t* p = baseT + (size_t)col * Kdim + k0 + kb;
    f.q[0] = *(const v4u*)p;
    f.q[1] = *(const v4u*)(p + 16);
}

// A fragment built on the fly from f32 (LDS) with hi/lo split
__device__ __forceinline__ void make_a_frag_f32(const float* A, int stride, int lane, int k0, BFrag& hi, BFrag& lo) {
    int r = lane & 15;
    int kb = (lane < 16) ? 0 : 8;
    const float* p = A + (size_t)r * stride + k0 + kb;
#pragma unroll
    for (int e = 0; e < 8; ++e) {
        float v = p[e];
        ushort_t h = f2bf(v);
        hi.u[e] = h;
        lo.u[e] = f2bf(v - bf2f(h));
    }
#pragma unroll
    for (int e = 0; e < 8; ++e) {
        float v = p[16 + e];
        ushort_t h = f2bf(v);
        hi.u[8 + e] = h;
        lo.u[8 + e] = f2bf(v - bf2f(h));
    }
}

// one 16x16 output tile: A = f32 [16 x Kdim] (LDS), B = bf16 hi/lo [N][Kdim]
__device__ __forceinline__ v8f gemm_f32A(const float* A, int aStride, int Kdim,
                                         const ushort_t* BT_hi, const ushort_t* BT_lo,
                                         int nBase, int lane) {
    v8f acc = zero8();
    for (int k0 = 0; k0 < Kdim; k0 += 32) {
        BFrag ah, al, bh, bl;
        make_a_frag_f32(A, aStride, lane, k0, ah, al);
        load_b_frag(BT_hi, Kdim, nBase, lane, k0, bh);
        load_b_frag(BT_lo, Kdim, nBase, lane, k0, bl);
        acc = wmma3(ah, al, bh, bl, acc);
    }
    return acc;
}

// LayerNorm over one row of 256 (one wave32 per row), in place
__device__ __forceinline__ void ln_row(float* row, int lane, const float* g, const float* be) {
    float vals[8];
    float s = 0.f, s2 = 0.f;
#pragma unroll
    for (int i = 0; i < 8; ++i) { float v = row[lane + 32 * i]; vals[i] = v; s += v; s2 += v * v; }
#pragma unroll
    for (int off = 16; off > 0; off >>= 1) { s += __shfl_xor(s, off); s2 += __shfl_xor(s2, off); }
    float mu = s * (1.f / 256.f);
    float var = s2 * (1.f / 256.f) - mu * mu;
    float rs = rsqrtf(var + 1e-5f);
#pragma unroll
    for (int i = 0; i < 8; ++i) { int c = lane + 32 * i; row[c] = (vals[i] - mu) * rs * g[c] + be[c]; }
}

// ---------------- kernels ----------------

// transpose weight [K][N] -> bf16 hi/lo [N][K]
__global__ __launch_bounds__(256) void k_prep(const float* __restrict__ w,
                                              ushort_t* __restrict__ hiT, ushort_t* __restrict__ loT,
                                              int K, int N) {
    int total = K * N;
    for (int i = blockIdx.x * 256 + threadIdx.x; i < total; i += gridDim.x * 256) {
        int k = i / N, n = i - k * N;
        float f = w[i];
        ushort_t h = f2bf(f);
        hiT[(size_t)n * K + k] = h;
        loT[(size_t)n * K + k] = f2bf(f - bf2f(h));
    }
}

// target = mean over T of hidden_states, phase 1.
// b128 non-temporal streaming loads: 512 MB at the HBM roofline without
// polluting the 192 MB L2 (the rest of the pipeline's working set is ~30 MB).
__global__ __launch_bounds__(256) void k_mean_part(const float* __restrict__ hidden, float* __restrict__ part) {
    int bid = blockIdx.x;        // 16 tc * 4 hc * 8 b = 512 blocks
    int tc = bid & 15;
    int hc = (bid >> 4) & 3;
    int b = bid >> 6;
    int h4 = hc * 1024 + threadIdx.x * 4;
    const float* base = hidden + ((size_t)b * 4096 + (size_t)tc * 256) * 4096 + h4;
    v4f acc = {0.f, 0.f, 0.f, 0.f};
    for (int t = 0; t < 256; ++t) {
        v4f v = __builtin_nontemporal_load((const v4f*)(base + (size_t)t * 4096));
        acc += v;
    }
    *(v4f*)(part + (size_t)tc * 32768 + b * 4096 + h4) = acc;
}
__global__ __launch_bounds__(256) void k_mean_fin(const float* __restrict__ part, float* __restrict__ target) {
    int i = blockIdx.x * 256 + threadIdx.x;
    float s = 0.f;
#pragma unroll
    for (int tc = 0; tc < 16; ++tc) s += part[(size_t)tc * 32768 + i];
    target[i] = s * (1.f / 4096.f);
}

// x = LN(gather(hidden) @ in_w + in_b); outputs x as bf16 hi/lo (row-major [4096][256]).
// A is staged in 128-wide K chunks (one float4 per thread, coalesced b128 loads),
// so each barrier pair covers 12 WMMAs instead of 3.
__global__ __launch_bounds__(512) void k_in_proj(
    const float* __restrict__ hidden,
    const ushort_t* __restrict__ inwT_hi, const ushort_t* __restrict__ inwT_lo,
    const float* __restrict__ in_b, const float* __restrict__ g, const float* __restrict__ be,
    ushort_t* __restrict__ x_hi, ushort_t* __restrict__ x_lo) {
    __shared__ alignas(16) ushort_t aHi[16][128];
    __shared__ alignas(16) ushort_t aLo[16][128];
    __shared__ float y[16][256];
    const int tile = blockIdx.x, tid = threadIdx.x, wave = tid >> 5, lane = tid & 31;
    // staging assignment: one float4 per thread per 128-K chunk
    const int sr = tid >> 5;   // row 0..15
    const int sk = (tid & 31) * 4;
    const int m = tile * 16 + sr;
    const int bb = m >> 9;
    const int nn = m & 511;
    const int tok = (nn < 256) ? (int)((float)nn * (3840.0f / 255.0f)) : (3840 + (nn - 256));
    const float* rowbase = hidden + ((size_t)bb * 4096 + (size_t)tok) * 4096;
    v8f acc = zero8();
    for (int kc = 0; kc < 4096; kc += 128) {
        v4f fv = *(const v4f*)(rowbase + kc + sk);
        __builtin_prefetch(rowbase + kc + 128 + sk, 0, 0);
#pragma unroll
        for (int i = 0; i < 4; ++i) {
            ushort_t hh = f2bf(fv[i]);
            aHi[sr][sk + i] = hh;
            aLo[sr][sk + i] = f2bf(fv[i] - bf2f(hh));
        }
        __syncthreads();
#pragma unroll
        for (int kk = 0; kk < 128; kk += 32) {
            BFrag ah, al, bh, bl;
            load_a_frag(&aHi[0][0], 128, lane, kk, ah);
            load_a_frag(&aLo[0][0], 128, lane, kk, al);
            load_b_frag(inwT_hi, 4096, wave * 16, lane, kc + kk, bh);
            load_b_frag(inwT_lo, 4096, wave * 16, lane, kc + kk, bl);
            acc = wmma3(ah, al, bh, bl, acc);
        }
        __syncthreads();
    }
    {
        int col = wave * 16 + (lane & 15);
        int rb = (lane < 16) ? 0 : 8;
#pragma unroll
        for (int j = 0; j < 8; ++j) y[rb + j][col] = acc[j] + in_b[col];
    }
    __syncthreads();
    {   // LN: wave handles row = wave
        float vals[8];
        float s = 0.f, s2 = 0.f;
#pragma unroll
        for (int i = 0; i < 8; ++i) { float v = y[wave][lane + 32 * i]; vals[i] = v; s += v; s2 += v * v; }
#pragma unroll
        for (int off = 16; off > 0; off >>= 1) { s += __shfl_xor(s, off); s2 += __shfl_xor(s2, off); }
        float mu = s * (1.f / 256.f);
        float var = s2 * (1.f / 256.f) - mu * mu;
        float rs = rsqrtf(var + 1e-5f);
        size_t rowOff = (size_t)(tile * 16 + wave) * 256;
#pragma unroll
        for (int i = 0; i < 8; ++i) {
            int c = lane + 32 * i;
            float xv = (vals[i] - mu) * rs * g[c] + be[c];
            ushort_t hh = f2bf(xv);
            x_hi[rowOff + c] = hh;
            x_lo[rowOff + c] = f2bf(xv - bf2f(hh));
        }
    }
}

// k = x@k_w (row layout [token][d]); v = x@v_w stored transposed [b][d][token]; bf16 hi/lo
__global__ __launch_bounds__(512) void k_kv(
    const ushort_t* __restrict__ x_hi, const ushort_t* __restrict__ x_lo,
    const ushort_t* __restrict__ kwT_hi, const ushort_t* __restrict__ kwT_lo,
    const ushort_t* __restrict__ vwT_hi, const ushort_t* __restrict__ vwT_lo,
    ushort_t* __restrict__ kmat_hi, ushort_t* __restrict__ kmat_lo,
    ushort_t* __restrict__ vT_hi, ushort_t* __restrict__ vT_lo) {
    const int tile = blockIdx.x, tid = threadIdx.x, wave = tid >> 5, lane = tid & 31;
    const ushort_t* aH = x_hi + (size_t)tile * 16 * 256;
    const ushort_t* aL = x_lo + (size_t)tile * 16 * 256;
    v8f ak = zero8(), av = zero8();
    for (int k0 = 0; k0 < 256; k0 += 32) {
        BFrag ah, al, bh, bl;
        load_a_frag(aH, 256, lane, k0, ah);
        load_a_frag(aL, 256, lane, k0, al);
        load_b_frag(kwT_hi, 256, wave * 16, lane, k0, bh);
        load_b_frag(kwT_lo, 256, wave * 16, lane, k0, bl);
        ak = wmma3(ah, al, bh, bl, ak);
        load_b_frag(vwT_hi, 256, wave * 16, lane, k0, bh);
        load_b_frag(vwT_lo, 256, wave * 16, lane, k0, bl);
        av = wmma3(ah, al, bh, bl, av);
    }
    int col = wave * 16 + (lane & 15);
    int rb = (lane < 16) ? 0 : 8;
#pragma unroll
    for (int j = 0; j < 8; ++j) {
        int mm = tile * 16 + rb + j;
        float kv = ak[j];
        ushort_t hh = f2bf(kv);
        kmat_hi[(size_t)mm * 256 + col] = hh;
        kmat_lo[(size_t)mm * 256 + col] = f2bf(kv - bf2f(hh));
        int bbb = mm >> 9, nnn = mm & 511;
        float vv = av[j];
        ushort_t vh = f2bf(vv);
        size_t vo = ((size_t)bbb * 256 + col) * 512 + nnn;
        vT_hi[vo] = vh;
        vT_lo[vo] = f2bf(vv - bf2f(vh));
    }
}

// 3 slot-attention iterations; one block per batch, 16 waves, all matmuls via WMMA
__global__ __launch_bounds__(512) void k_slots(
    const float* __restrict__ old_slots,
    const ushort_t* __restrict__ qwT_hi, const ushort_t* __restrict__ qwT_lo,
    const ushort_t* __restrict__ kmat_hi, const ushort_t* __restrict__ kmat_lo,
    const ushort_t* __restrict__ vT_hi, const ushort_t* __restrict__ vT_lo,
    const ushort_t* __restrict__ w1T_hi, const ushort_t* __restrict__ w1T_lo,
    const ushort_t* __restrict__ w2T_hi, const ushort_t* __restrict__ w2T_lo,
    const float* __restrict__ ln_s_g, const float* __restrict__ ln_s_b,
    const float* __restrict__ ln_m_g, const float* __restrict__ ln_m_b,
    const float* __restrict__ mlp_b1, const float* __restrict__ mlp_b2,
    float* __restrict__ out_slots) {
    __shared__ alignas(16) float SLOT[16][256];
    __shared__ alignas(16) float QB[16][256];
    __shared__ alignas(16) float BIG[16][512];
    const int b = blockIdx.x, tid = threadIdx.x, wave = tid >> 5, lane = tid & 31;
    const ushort_t* kh = kmat_hi + (size_t)b * 512 * 256;
    const ushort_t* kl = kmat_lo + (size_t)b * 512 * 256;
    const ushort_t* vh = vT_hi + (size_t)b * 256 * 512;
    const ushort_t* vl = vT_lo + (size_t)b * 256 * 512;

    for (int i = tid; i < 4096; i += 512) SLOT[i >> 8][i & 255] = old_slots[(size_t)b * 4096 + i];
    __syncthreads();

    for (int it = 0; it < 3; ++it) {
        ln_row(&SLOT[wave][0], lane, ln_s_g, ln_s_b);  // SLOT <- s = LN(slots)
        __syncthreads();
        {   // q = s @ q_w
            v8f acc = gemm_f32A(&SLOT[0][0], 256, 256, qwT_hi, qwT_lo, wave * 16, lane);
            int col = wave * 16 + (lane & 15);
            int rb = (lane < 16) ? 0 : 8;
#pragma unroll
            for (int j = 0; j < 8; ++j) QB[rb + j][col] = acc[j];
        }
        __syncthreads();
#pragma unroll
        for (int t2 = 0; t2 < 2; ++t2) {  // logits = (q @ k^T) * D^-0.5
            int nb = (wave * 2 + t2) * 16;
            v8f acc = gemm_f32A(&QB[0][0], 256, 256, kh, kl, nb, lane);
            int col = nb + (lane & 15);
            int rb = (lane < 16) ? 0 : 8;
#pragma unroll
            for (int j = 0; j < 8; ++j) BIG[rb + j][col] = acc[j] * 0.0625f;
        }
        __syncthreads();
        {   // softmax over slot axis (16) per token; one thread per token
            int n = tid;
            float mx = BIG[0][n];
#pragma unroll
            for (int k = 1; k < 16; ++k) mx = fmaxf(mx, BIG[k][n]);
            float e[16];
            float s = 0.f;
#pragma unroll
            for (int k = 0; k < 16; ++k) { e[k] = expf(BIG[k][n] - mx); s += e[k]; }
            float inv = 1.f / s;
#pragma unroll
            for (int k = 0; k < 16; ++k) BIG[k][n] = e[k] * inv;
        }
        __syncthreads();
        {   // upd = attn @ v
            v8f acc = gemm_f32A(&BIG[0][0], 512, 512, vh, vl, wave * 16, lane);
            int col = wave * 16 + (lane & 15);
            int rb = (lane < 16) ? 0 : 8;
#pragma unroll
            for (int j = 0; j < 8; ++j) QB[rb + j][col] = acc[j];
        }
        __syncthreads();
        ln_row(&QB[wave][0], lane, ln_m_g, ln_m_b);  // h = LN(upd)
        __syncthreads();
#pragma unroll
        for (int t2 = 0; t2 < 2; ++t2) {  // t1 = gelu(h @ w1 + b1)
            int nb = (wave * 2 + t2) * 16;
            v8f acc = gemm_f32A(&QB[0][0], 256, 256, w1T_hi, w1T_lo, nb, lane);
            int col = nb + (lane & 15);
            int rb = (lane < 16) ? 0 : 8;
#pragma unroll
            for (int j = 0; j < 8; ++j) BIG[rb + j][col] = gelu(acc[j] + mlp_b1[col]);
        }
        __syncthreads();
        {   // slots = s + t1 @ w2 + b2
            v8f acc = gemm_f32A(&BIG[0][0], 512, 512, w2T_hi, w2T_lo, wave * 16, lane);
            int col = wave * 16 + (lane & 15);
            int rb = (lane < 16) ? 0 : 8;
#pragma unroll
            for (int j = 0; j < 8; ++j) SLOT[rb + j][col] += acc[j] + mlp_b2[col];
        }
        __syncthreads();
    }
    for (int i = tid; i < 4096; i += 512) out_slots[(size_t)b * 4096 + i] = SLOT[i >> 8][i & 255];
}

// decoder + per-batch loss partial; one block per batch
__global__ __launch_bounds__(512) void k_decoder(
    const float* __restrict__ slots,
    const ushort_t* __restrict__ d1T_hi, const ushort_t* __restrict__ d1T_lo,
    const ushort_t* __restrict__ d2T_hi, const ushort_t* __restrict__ d2T_lo,
    const ushort_t* __restrict__ d3T_hi, const ushort_t* __restrict__ d3T_lo,
    const float* __restrict__ b1, const float* __restrict__ b2, const float* __restrict__ b3,
    const float* __restrict__ target, float* __restrict__ lossPart) {
    __shared__ alignas(16) float ACT[16][512];
    __shared__ float red[16];
    const int b = blockIdx.x, tid = threadIdx.x, wave = tid >> 5, lane = tid & 31;
    for (int i = tid; i < 4096; i += 512) ACT[i >> 8][i & 255] = slots[(size_t)b * 4096 + i];
    __syncthreads();
    {   // stage1: 16x512 = slots(16x256) @ d1
        int nb0 = wave * 32, nb1 = wave * 32 + 16;
        v8f a0 = gemm_f32A(&ACT[0][0], 512, 256, d1T_hi, d1T_lo, nb0, lane);
        v8f a1 = gemm_f32A(&ACT[0][0], 512, 256, d1T_hi, d1T_lo, nb1, lane);
        __syncthreads();
        int c0 = nb0 + (lane & 15), c1 = nb1 + (lane & 15);
        int rb = (lane < 16) ? 0 : 8;
#pragma unroll
        for (int j = 0; j < 8; ++j) {
            ACT[rb + j][c0] = gelu(a0[j] + b1[c0]);
            ACT[rb + j][c1] = gelu(a1[j] + b1[c1]);
        }
    }
    __syncthreads();
    {   // stage2: 16x512 = r1 @ d2
        int nb0 = wave * 32, nb1 = wave * 32 + 16;
        v8f a0 = gemm_f32A(&ACT[0][0], 512, 512, d2T_hi, d2T_lo, nb0, lane);
        v8f a1 = gemm_f32A(&ACT[0][0], 512, 512, d2T_hi, d2T_lo, nb1, lane);
        __syncthreads();
        int c0 = nb0 + (lane & 15), c1 = nb1 + (lane & 15);
        int rb = (lane < 16) ? 0 : 8;
#pragma unroll
        for (int j = 0; j < 8; ++j) {
            ACT[rb + j][c0] = gelu(a0[j] + b2[c0]);
            ACT[rb + j][c1] = gelu(a1[j] + b2[c1]);
        }
    }
    __syncthreads();
    // stage3: r2 @ d3 (N=4096), fused mean-over-16-slots + squared error
    float lossAcc = 0.f;
    for (int t = 0; t < 16; ++t) {
        int nb = (wave * 16 + t) * 16;
        v8f acc = gemm_f32A(&ACT[0][0], 512, 512, d3T_hi, d3T_lo, nb, lane);
        float cs = 0.f;
#pragma unroll
        for (int j = 0; j < 8; ++j) cs += acc[j];  // rows rb..rb+7
        cs += __shfl_xor(cs, 16);                  // + other 8 rows (same col in lane^16)
        if (lane < 16) {
            int col = nb + lane;
            float recon = cs * (1.f / 16.f) + b3[col];
            float d = recon - target[b * 4096 + col];
            lossAcc += d * d;
        }
    }
#pragma unroll
    for (int off = 16; off > 0; off >>= 1) lossAcc += __shfl_xor(lossAcc, off);
    if (lane == 0) red[wave] = lossAcc;
    __syncthreads();
    if (tid == 0) {
        float s = 0.f;
#pragma unroll
        for (int w = 0; w < 16; ++w) s += red[w];
        lossPart[b] = s;
    }
}

__global__ void k_loss_fin(const float* __restrict__ lp, float* __restrict__ out) {
    if (threadIdx.x == 0) {
        float s = 0.f;
        for (int i = 0; i < 8; ++i) s += lp[i];
        *out = s * (1.f / 32768.f);
    }
}

extern "C" void kernel_launch(void* const* d_in, const int* in_sizes, int n_in,
                              void* d_out, int out_size, void* d_ws, size_t ws_size,
                              hipStream_t stream) {
    (void)in_sizes; (void)n_in; (void)out_size; (void)ws_size;
    const float* hidden    = (const float*)d_in[0];
    const float* old_slots = (const float*)d_in[1];
    const float* in_w   = (const float*)d_in[2];
    const float* in_b   = (const float*)d_in[3];
    const float* ln_in_g = (const float*)d_in[4];
    const float* ln_in_b = (const float*)d_in[5];
    const float* ln_s_g = (const float*)d_in[6];
    const float* ln_s_b = (const float*)d_in[7];
    const float* ln_m_g = (const float*)d_in[8];
    const float* ln_m_b = (const float*)d_in[9];
    const float* q_w    = (const float*)d_in[10];
    const float* k_w    = (const float*)d_in[11];
    const float* v_w    = (const float*)d_in[12];
    const float* mlp_w1 = (const float*)d_in[13];
    const float* mlp_b1 = (const float*)d_in[14];
    const float* mlp_w2 = (const float*)d_in[15];
    const float* mlp_b2 = (const float*)d_in[16];
    const float* dec_w1 = (const float*)d_in[17];
    const float* dec_b1 = (const float*)d_in[18];
    const float* dec_w2 = (const float*)d_in[19];
    const float* dec_b2 = (const float*)d_in[20];
    const float* dec_w3 = (const float*)d_in[21];
    const float* dec_b3 = (const float*)d_in[22];
    float* out = (float*)d_out;

    char* p = (char*)d_ws;
    auto alloc = [&](size_t bytes) -> char* {
        char* r = p;
        p += (bytes + 255) & ~(size_t)255;
        return r;
    };
    float* target    = (float*)alloc(32768u * 4);
    float* meanPart  = (float*)alloc(16u * 32768u * 4);
    float* lossPart  = (float*)alloc(8u * 4);
    ushort_t* x_hi   = (ushort_t*)alloc(4096u * 256u * 2);
    ushort_t* x_lo   = (ushort_t*)alloc(4096u * 256u * 2);
    ushort_t* kmat_hi = (ushort_t*)alloc(4096u * 256u * 2);
    ushort_t* kmat_lo = (ushort_t*)alloc(4096u * 256u * 2);
    ushort_t* vT_hi  = (ushort_t*)alloc(4096u * 256u * 2);
    ushort_t* vT_lo  = (ushort_t*)alloc(4096u * 256u * 2);
    ushort_t* inwT_hi = (ushort_t*)alloc(4096u * 256u * 2);
    ushort_t* inwT_lo = (ushort_t*)alloc(4096u * 256u * 2);
    ushort_t* qwT_hi = (ushort_t*)alloc(256u * 256u * 2);
    ushort_t* qwT_lo = (ushort_t*)alloc(256u * 256u * 2);
    ushort_t* kwT_hi = (ushort_t*)alloc(256u * 256u * 2);
    ushort_t* kwT_lo = (ushort_t*)alloc(256u * 256u * 2);
    ushort_t* vwT_hi = (ushort_t*)alloc(256u * 256u * 2);
    ushort_t* vwT_lo = (ushort_t*)alloc(256u * 256u * 2);
    ushort_t* w1T_hi = (ushort_t*)alloc(256u * 512u * 2);
    ushort_t* w1T_lo = (ushort_t*)alloc(256u * 512u * 2);
    ushort_t* w2T_hi = (ushort_t*)alloc(512u * 256u * 2);
    ushort_t* w2T_lo = (ushort_t*)alloc(512u * 256u * 2);
    ushort_t* d1T_hi = (ushort_t*)alloc(256u * 512u * 2);
    ushort_t* d1T_lo = (ushort_t*)alloc(256u * 512u * 2);
    ushort_t* d2T_hi = (ushort_t*)alloc(512u * 512u * 2);
    ushort_t* d2T_lo = (ushort_t*)alloc(512u * 512u * 2);
    ushort_t* d3T_hi = (ushort_t*)alloc(512u * 4096u * 2);
    ushort_t* d3T_lo = (ushort_t*)alloc(512u * 4096u * 2);

    // weight prep (transpose + bf16 hi/lo split)
    k_prep<<<4096, 256, 0, stream>>>(in_w, inwT_hi, inwT_lo, 4096, 256);
    k_prep<<<256, 256, 0, stream>>>(q_w, qwT_hi, qwT_lo, 256, 256);
    k_prep<<<256, 256, 0, stream>>>(k_w, kwT_hi, kwT_lo, 256, 256);
    k_prep<<<256, 256, 0, stream>>>(v_w, vwT_hi, vwT_lo, 256, 256);
    k_prep<<<512, 256, 0, stream>>>(mlp_w1, w1T_hi, w1T_lo, 256, 512);
    k_prep<<<512, 256, 0, stream>>>(mlp_w2, w2T_hi, w2T_lo, 512, 256);
    k_prep<<<512, 256, 0, stream>>>(dec_w1, d1T_hi, d1T_lo, 256, 512);
    k_prep<<<1024, 256, 0, stream>>>(dec_w2, d2T_hi, d2T_lo, 512, 512);
    k_prep<<<8192, 256, 0, stream>>>(dec_w3, d3T_hi, d3T_lo, 512, 4096);

    // target = mean(hidden, axis=1): streaming b128 NT loads, 512 MB @ HBM roofline
    k_mean_part<<<512, 256, 0, stream>>>(hidden, meanPart);
    k_mean_fin<<<128, 256, 0, stream>>>(meanPart, target);

    // x = LN(gather @ in_w + b); then k/v projections
    k_in_proj<<<256, 512, 0, stream>>>(hidden, inwT_hi, inwT_lo, in_b, ln_in_g, ln_in_b, x_hi, x_lo);
    k_kv<<<256, 512, 0, stream>>>(x_hi, x_lo, kwT_hi, kwT_lo, vwT_hi, vwT_lo,
                                  kmat_hi, kmat_lo, vT_hi, vT_lo);

    // 3 slot-attention iterations (writes slots to d_out[0:32768])
    k_slots<<<8, 512, 0, stream>>>(old_slots, qwT_hi, qwT_lo, kmat_hi, kmat_lo, vT_hi, vT_lo,
                                   w1T_hi, w1T_lo, w2T_hi, w2T_lo,
                                   ln_s_g, ln_s_b, ln_m_g, ln_m_b, mlp_b1, mlp_b2, out);

    // decoder + loss
    k_decoder<<<8, 512, 0, stream>>>(out, d1T_hi, d1T_lo, d2T_hi, d2T_lo, d3T_hi, d3T_lo,
                                     dec_b1, dec_b2, dec_b3, target, lossPart);
    k_loss_fin<<<1, 32, 0, stream>>>(lossPart, out + 32768);
}
